// MultiHeadAttention_3676492005605
// MI455X (gfx1250) — matrix operand order repaired and checked by reading, not compiled, not run on hardware
//
#include <hip/hip_runtime.h>

// ---------------------------------------------------------------------------
// MI455X (gfx1250, wave32) low-rank multi-head attention.
//   proj GEMMs (bf16 WMMA, fp32 accum, double-buffered LDS)  +
//   flash attention (bf16 WMMA, online softmax).
// All global intermediates fp32 (28MB total -> resident in 192MB L2);
// bf16 conversion at LDS-staging time via packed v_cvt_pk_bf16_f32.
// WMMA shape: v_wmma_f32_16x16x32_bf16 (16384 FLOP / instruction / wave).
// ---------------------------------------------------------------------------

typedef __attribute__((ext_vector_type(16))) __bf16          v16bf;
typedef __attribute__((ext_vector_type(8)))  float           v8f;
typedef __attribute__((ext_vector_type(4)))  float           f4;
typedef __attribute__((ext_vector_type(8)))  unsigned short  v8us;
typedef __attribute__((ext_vector_type(16))) unsigned short  v16us;
typedef __attribute__((ext_vector_type(2)))  unsigned int    v2u;

#define SEQ   4096
#define DM    1024
#define HEADS 8
#define ZIP   256
#define DHQ   32     // ZIP / HEADS
#define DHV   128    // DM  / HEADS

#define QK_PITCH 40   // 32 k + 8 pad halfwords: 80B rows, 16B aligned, conflict-free
#define PV_PITCH 136  // 128 k + 8 pad halfwords: 272B rows, 16B aligned, conflict-free

// two fp32 -> packed 2x bf16 in one dword (one v_cvt_pk_bf16_f32)
__device__ __forceinline__ unsigned int pk2bf(float x, float y) {
#if __has_builtin(__builtin_amdgcn_cvt_pk_bf16_f32)
  auto r = __builtin_amdgcn_cvt_pk_bf16_f32(x, y);
  union { decltype(r) h; unsigned int u; } c;
  c.h = r;
  return c.u;
#elif defined(__gfx1250__)
  unsigned int r;
  asm("v_cvt_pk_bf16_f32 %0, %1, %2" : "=v"(r) : "v"(x), "v"(y));
  return r;
#else
  union { __bf16 h[2]; unsigned int u; } c;
  c.h[0] = (__bf16)x;
  c.h[1] = (__bf16)y;
  return c.u;
#endif
}

// fp32 -> bf16 in VGPR[15:0]; pairs with ds_store_b16 (no extra pack ops)
__device__ __forceinline__ unsigned short f2bf(float f) {
  return (unsigned short)pk2bf(f, f);
}

// pack 4 floats -> 4 bf16, one ds_store_b64
__device__ __forceinline__ void st_bf16x4(unsigned short* p, f4 v) {
  v2u d;
  d.x = pk2bf(v.x, v.y);
  d.y = pk2bf(v.z, v.w);
  *(v2u*)p = d;
}

// two contiguous 16B LDS loads -> one 16-element bf16 fragment
__device__ __forceinline__ v16bf frag_ld(const unsigned short* p0,
                                         const unsigned short* p1) {
  v8us lo = *(const v8us*)p0;
  v8us hi = *(const v8us*)p1;
  union { v16us u; v16bf h; } f;
#pragma unroll
  for (int i = 0; i < 8; ++i) { f.u[i] = lo[i]; f.u[i + 8] = hi[i]; }
  return f.h;
}

__device__ __forceinline__ v8f wmma_bf16(v16bf a, v16bf b, v8f c) {
  // (neg_a, A, neg_b, B, c_mod, C, reuse_a, reuse_b)
  return __builtin_amdgcn_wmma_f32_16x16x32_bf16(false, a, false, b,
                                                 (short)0, c, false, false);
}

__device__ __forceinline__ v8f zero8() {
  v8f z;
#pragma unroll
  for (int i = 0; i < 8; ++i) z[i] = 0.0f;
  return z;
}

// ---------------------------------------------------------------------------
// C[M,N] = A[M,K] @ W[N,K]^T + bias[N]      (torch Linear layout)
// grid (M/128, N/128), block 256 (8 waves). Wave w computes a 32x64 sub-tile.
// Double-buffered LDS: slab k+1 is loaded to registers before the WMMA burst
// of slab k and written to the alternate LDS buffer after it, so global-load
// latency hides under matrix math. One barrier per slab.
// ---------------------------------------------------------------------------
__global__ __launch_bounds__(256) void gemm_nt_bias(
    const float* __restrict__ A, const float* __restrict__ W,
    const float* __restrict__ bias, float* __restrict__ C,
    int M, int N, int K) {
  __shared__ unsigned short As[2][128 * QK_PITCH];
  __shared__ unsigned short Ws[2][128 * QK_PITCH];

  const int tid  = threadIdx.x;
  const int lane = tid & 31, wave = tid >> 5;
  const int m0g  = blockIdx.x * 128, n0g = blockIdx.y * 128;
  const int mw   = (wave & 3) * 32,  nw  = (wave >> 2) * 64;
  const int half = lane >> 4;         // 0: lanes 0-15, 1: lanes 16-31
  const int lcol = lane & 15;
  const int ka   = half ? 8 : 0;      // A-fragment k-slab base (ISA 16-bit A layout)
  const int kb   = half ? 8 : 0;      // B-fragment k-slab base (same k order as A)
  const int srow = tid >> 3;          // staging row base (0..31)
  const int scol = (tid & 7) * 4;     // staging col (0,4,...,28)

  // ---- stage slab 0 into buffer 0
#pragma unroll
  for (int it = 0; it < 4; ++it) {
    int r = srow + it * 32;
    f4 a = *(const f4*)(A + (long)(m0g + r) * K + scol);
    f4 w = *(const f4*)(W + (long)(n0g + r) * K + scol);
    st_bf16x4(&As[0][r * QK_PITCH + scol], a);
    st_bf16x4(&Ws[0][r * QK_PITCH + scol], w);
  }

  v8f acc[2][4];
#pragma unroll
  for (int i = 0; i < 2; ++i)
#pragma unroll
    for (int j = 0; j < 4; ++j) acc[i][j] = zero8();

  const int nk = K / 32;
  int buf = 0;
  for (int ks = 0; ks < nk; ++ks) {
    __syncthreads();  // staged slab `buf` visible; all reads of buf^1 done

    // ---- issue global loads for slab ks+1 (latency overlaps the WMMAs)
    f4 an[4], wn[4];
    const bool more = (ks + 1 < nk);
    if (more) {
      const int k0 = (ks + 1) * 32;
#pragma unroll
      for (int it = 0; it < 4; ++it) {
        int r = srow + it * 32;
        an[it] = *(const f4*)(A + (long)(m0g + r) * K + k0 + scol);
        wn[it] = *(const f4*)(W + (long)(n0g + r) * K + k0 + scol);
      }
    }

    // ---- WMMA burst on slab `buf`
    const unsigned short* Ab = As[buf];
    const unsigned short* Wb = Ws[buf];
    const unsigned short* a0p = &Ab[(mw + lcol) * QK_PITCH + ka];
    const unsigned short* a1p = &Ab[(mw + 16 + lcol) * QK_PITCH + ka];
    v16bf a0 = frag_ld(a0p, a0p + 16);
    v16bf a1 = frag_ld(a1p, a1p + 16);
    v16bf bfr[4];
#pragma unroll
    for (int n = 0; n < 4; ++n) {
      const unsigned short* bp = &Wb[(nw + n * 16 + lcol) * QK_PITCH + kb];
      bfr[n] = frag_ld(bp, bp + 16);
    }
#pragma unroll
    for (int n = 0; n < 4; ++n) {
      acc[0][n] = wmma_bf16(a0, bfr[n], acc[0][n]);
      acc[1][n] = wmma_bf16(a1, bfr[n], acc[1][n]);
    }

    // ---- convert and store slab ks+1 into the alternate buffer
    if (more) {
#pragma unroll
      for (int it = 0; it < 4; ++it) {
        int r = srow + it * 32;
        st_bf16x4(&As[buf ^ 1][r * QK_PITCH + scol], an[it]);
        st_bf16x4(&Ws[buf ^ 1][r * QK_PITCH + scol], wn[it]);
      }
    }
    buf ^= 1;
  }

#pragma unroll
  for (int n = 0; n < 4; ++n) {
    int col = n0g + nw + n * 16 + lcol;
    float b = bias[col];
#pragma unroll
    for (int mt = 0; mt < 2; ++mt)
#pragma unroll
      for (int r = 0; r < 8; ++r) {
        int row = m0g + mw + mt * 16 + half * 8 + r;   // ISA C/D layout
        C[(long)row * N + col] = acc[mt][n][r] + b;
      }
  }
}

// ---------------------------------------------------------------------------
// Flash attention, no 1/sqrt(dk) scaling (matches reference).
// grid (SEQ/128, HEADS), block 256. Wave w owns q-rows [16w,16w+16): softmax
// row statistics stay inside one wave (reduced over 16-lane halves, matching
// the WMMA C layout row = VGPR_index + 8*(lane>=16)).
// ---------------------------------------------------------------------------
__global__ __launch_bounds__(256) void flash_attn(
    const float* __restrict__ Qp,   // [SEQ, ZIP]
    const float* __restrict__ Kp,   // [SEQ, ZIP]
    const float* __restrict__ Vp,   // [SEQ, DM]
    float* __restrict__ Out) {      // [SEQ, DM]
  __shared__ unsigned short Qs[128 * QK_PITCH];   // 128 x 32  bf16
  __shared__ unsigned short Ks[128 * QK_PITCH];   // 128 x 32  bf16
  __shared__ unsigned short Vt[128 * PV_PITCH];   // transposed: [c][keyrow]
  __shared__ unsigned short Ps[128 * PV_PITCH];   // P tile, per-wave 16-row slabs

  const int tid  = threadIdx.x;
  const int lane = tid & 31, wave = tid >> 5;
  const int qb   = blockIdx.x, h = blockIdx.y;
  const int half = lane >> 4, lcol = lane & 15;
  const int ka   = half ? 8 : 0;
  const int kb   = half ? 8 : 0;

  // ---- stage Q tile (128 x 32) once
  {
    int r  = tid >> 1;
    int c0 = (tid & 1) * 16;
#pragma unroll
    for (int i = 0; i < 4; ++i) {
      int c = c0 + i * 4;
      f4 qv = *(const f4*)(Qp + (long)(qb * 128 + r) * ZIP + h * DHQ + c);
      st_bf16x4(&Qs[r * QK_PITCH + c], qv);
    }
  }

  float mrun[8], lrun[8];
  v8f O[8];
#pragma unroll
  for (int r = 0; r < 8; ++r) { mrun[r] = -3.0e38f; lrun[r] = 0.0f; }
#pragma unroll
  for (int t = 0; t < 8; ++t) O[t] = zero8();

  for (int j = 0; j < SEQ / 128; ++j) {
    __syncthreads();  // previous tile consumed (also publishes Q on j==0)

    // ---- stage K tile (128 x 32)
    {
      int r  = tid >> 1;
      int c0 = (tid & 1) * 16;
#pragma unroll
      for (int i = 0; i < 4; ++i) {
        int c = c0 + i * 4;
        f4 kv = *(const f4*)(Kp + (long)(j * 128 + r) * ZIP + h * DHQ + c);
        st_bf16x4(&Ks[r * QK_PITCH + c], kv);
      }
    }
    // ---- stage V tile transposed, packing key-row pairs into b32 stores:
    //      Vt[c][2rp..2rp+1] = V[j*128 + 2rp..2rp+1][h*128+c]
#pragma unroll
    for (int it = 0; it < 8; ++it) {
      int lin = tid + 256 * it;           // 2048 (rowpair, col4) slots
      int rp  = lin >> 5;                 // 0..63
      int c4  = (lin & 31) * 4;
      f4 v0 = *(const f4*)(Vp + (long)(j * 128 + 2 * rp + 0) * DM + h * DHV + c4);
      f4 v1 = *(const f4*)(Vp + (long)(j * 128 + 2 * rp + 1) * DM + h * DHV + c4);
#pragma unroll
      for (int i = 0; i < 4; ++i)
        *(unsigned int*)&Vt[(c4 + i) * PV_PITCH + 2 * rp] = pk2bf(v0[i], v1[i]);
    }
    // prefetch next K/V tiles into cache (global_prefetch_b8)
    if (j + 1 < SEQ / 128) {
      __builtin_prefetch(Kp + (long)((j + 1) * 128 + (tid >> 1)) * ZIP + h * DHQ, 0, 1);
      __builtin_prefetch(Vp + (long)((j + 1) * 128 + (tid >> 1)) * DM + h * DHV, 0, 1);
    }
    __syncthreads();

    // ---- S = Q K^T  (16 q-rows x 128 keys per wave; K dim = 32 = one WMMA)
    const unsigned short* qp = &Qs[(wave * 16 + lcol) * QK_PITCH + ka];
    v16bf qf = frag_ld(qp, qp + 16);
    v16bf kfr[8];
#pragma unroll
    for (int t = 0; t < 8; ++t) {
      const unsigned short* kp = &Ks[(t * 16 + lcol) * QK_PITCH + kb];
      kfr[t] = frag_ld(kp, kp + 16);
    }
    v8f S[8];
#pragma unroll
    for (int t = 0; t < 8; ++t) S[t] = wmma_bf16(qf, kfr[t], zero8());

    // ---- online softmax (row = half*8 + r; reduce across the 16-lane half)
#pragma unroll
    for (int r = 0; r < 8; ++r) {
      float mx = S[0][r];
#pragma unroll
      for (int t = 1; t < 8; ++t) mx = fmaxf(mx, S[t][r]);
#pragma unroll
      for (int off = 1; off < 16; off <<= 1) mx = fmaxf(mx, __shfl_xor(mx, off, 16));
      float mnew  = fmaxf(mrun[r], mx);
      float alpha = __expf(mrun[r] - mnew);
      float rs = 0.0f;
#pragma unroll
      for (int t = 0; t < 8; ++t) {
        float p = __expf(S[t][r] - mnew);
        S[t][r] = p;
        rs += p;
      }
#pragma unroll
      for (int off = 1; off < 16; off <<= 1) rs += __shfl_xor(rs, off, 16);
      lrun[r] = lrun[r] * alpha + rs;
      mrun[r] = mnew;
#pragma unroll
      for (int t = 0; t < 8; ++t) O[t][r] *= alpha;
    }

    // ---- spill P (bf16) to this wave's private LDS slab (same-wave DS order)
#pragma unroll
    for (int t = 0; t < 8; ++t)
#pragma unroll
      for (int r = 0; r < 8; ++r)
        Ps[(wave * 16 + half * 8 + r) * PV_PITCH + t * 16 + lcol] = f2bf(S[t][r]);

    // ---- O += P @ V   (4 k-slabs x 8 col-tiles = 32 WMMAs per wave)
#pragma unroll
    for (int kk = 0; kk < 4; ++kk) {
      const unsigned short* pp = &Ps[(wave * 16 + lcol) * PV_PITCH + kk * 32 + ka];
      v16bf pf = frag_ld(pp, pp + 16);
      v16bf vfr[8];
#pragma unroll
      for (int t = 0; t < 8; ++t) {
        const unsigned short* vp = &Vt[(t * 16 + lcol) * PV_PITCH + kk * 32 + kb];
        vfr[t] = frag_ld(vp, vp + 16);
      }
#pragma unroll
      for (int t = 0; t < 8; ++t) O[t] = wmma_bf16(pf, vfr[t], O[t]);
    }
  }

  // ---- epilogue: normalize and store
#pragma unroll
  for (int t = 0; t < 8; ++t)
#pragma unroll
    for (int r = 0; r < 8; ++r) {
      int row = qb * 128 + wave * 16 + half * 8 + r;
      int col = h * DHV + t * 16 + lcol;
      Out[(long)row * DM + col] = O[t][r] / lrun[r];
    }
}

// ---------------------------------------------------------------------------
extern "C" void kernel_launch(void* const* d_in, const int* in_sizes, int n_in,
                              void* d_out, int out_size, void* d_ws, size_t ws_size,
                              hipStream_t stream) {
  const float* q     = (const float*)d_in[0];
  const float* k     = (const float*)d_in[1];
  const float* v     = (const float*)d_in[2];
  const float* w_q   = (const float*)d_in[3];
  const float* b_q   = (const float*)d_in[4];
  const float* w_k   = (const float*)d_in[5];
  const float* b_k   = (const float*)d_in[6];
  const float* w_v_r = (const float*)d_in[7];
  const float* b_v_r = (const float*)d_in[8];
  const float* w_v_l = (const float*)d_in[9];
  const float* b_v_l = (const float*)d_in[10];

  float* query = (float*)d_ws;              // [SEQ, ZIP]
  float* keyp  = query + (size_t)SEQ * ZIP; // [SEQ, ZIP]
  float* tmpv  = keyp  + (size_t)SEQ * ZIP; // [SEQ, ZIP]
  float* value = tmpv  + (size_t)SEQ * ZIP; // [SEQ, DM]

  dim3 blk(256);
  gemm_nt_bias<<<dim3(SEQ / 128, ZIP / 128), blk, 0, stream>>>(q, w_q, b_q, query, SEQ, ZIP, DM);
  gemm_nt_bias<<<dim3(SEQ / 128, ZIP / 128), blk, 0, stream>>>(k, w_k, b_k, keyp, SEQ, ZIP, DM);
  gemm_nt_bias<<<dim3(SEQ / 128, ZIP / 128), blk, 0, stream>>>(v, w_v_r, b_v_r, tmpv, SEQ, ZIP, DM);
  gemm_nt_bias<<<dim3(SEQ / 128, DM / 128),  blk, 0, stream>>>(tmpv, w_v_l, b_v_l, value, SEQ, DM, ZIP);
  flash_attn<<<dim3(SEQ / 128, HEADS), blk, 0, stream>>>(query, keyp, value, (float*)d_out);
}
